// CrossAttention_71287867179098
// MI455X (gfx1250) — matrix operand order repaired and checked by reading, not compiled, not run on hardware
//
#include <hip/hip_runtime.h>
#include <hip/hip_bf16.h>
#include <stdint.h>

// ---------------------------------------------------------------------------
// Types for CDNA5 WMMA (wave32): v_wmma_f32_16x16x32_bf16
// ---------------------------------------------------------------------------
typedef __bf16 bf16_t;
typedef bf16_t v16bf __attribute__((ext_vector_type(16)));
typedef float  v8f   __attribute__((ext_vector_type(8)));
typedef unsigned int u32x4 __attribute__((ext_vector_type(4)));
typedef int          i32x4 __attribute__((ext_vector_type(4)));

union AB32 {            // one A or B fragment: 16 bf16 per lane = 32 bytes
    v16bf v;
    u32x4 q[2];
};

__device__ __forceinline__ uint16_t f2bf(float f) {
    union { float f; uint32_t u; } x; x.f = f;
    uint32_t u = x.u + 0x7FFFu + ((x.u >> 16) & 1u);   // round-to-nearest-even
    return (uint16_t)(u >> 16);
}

// Packed f32x2 -> bf16x2: use the CDNA5 hardware convert when available.
#if defined(__has_builtin)
#if __has_builtin(__builtin_amdgcn_cvt_pk_bf16_f32)
#define HAVE_PK_BF16 1
#endif
#endif

__device__ __forceinline__ uint32_t f2bf2(float lo, float hi) {
#ifdef HAVE_PK_BF16
    auto r = __builtin_amdgcn_cvt_pk_bf16_f32(lo, hi);
    union { decltype(r) v; uint32_t u; } cv; cv.v = r;
    return cv.u;
#else
    return (uint32_t)f2bf(lo) | ((uint32_t)f2bf(hi) << 16);
#endif
}

#define WMMA_BF16(A, B, C) \
    __builtin_amdgcn_wmma_f32_16x16x32_bf16(false, (A), false, (B), (short)0, (C), false, false)

// ---------------------------------------------------------------------------
// CDNA5 async global->LDS copy (ASYNCcnt path), with safe fallback.
// Builtin signature (probe-confirmed): (v4i AS1*, v4i AS3*, Ii off, Ii cpol)
// ---------------------------------------------------------------------------
#if defined(__has_builtin)
#if __has_builtin(__builtin_amdgcn_global_load_async_to_lds_b128)
#define HAVE_ASYNC_LDS 1
#endif
#endif

#ifdef HAVE_ASYNC_LDS
typedef __attribute__((address_space(1))) i32x4* as1_v4i_ptr;
typedef __attribute__((address_space(3))) i32x4* as3_v4i_ptr;
#endif

__device__ __forceinline__ void cp_async_b128(const uint16_t* g, uint16_t* l) {
#ifdef HAVE_ASYNC_LDS
    __builtin_amdgcn_global_load_async_to_lds_b128(
        (as1_v4i_ptr)(i32x4*)(void*)(uintptr_t)g,
        (as3_v4i_ptr)(i32x4*)(void*)l, 0, 0);
#else
    *(u32x4*)l = *(const u32x4*)g;
#endif
}

__device__ __forceinline__ void wait_async0() {
#ifdef HAVE_ASYNC_LDS
#if __has_builtin(__builtin_amdgcn_s_wait_asynccnt)
    __builtin_amdgcn_s_wait_asynccnt(0);
#else
    asm volatile("s_wait_asynccnt 0x0" ::: "memory");
#endif
#endif
}

// ---------------------------------------------------------------------------
// Kernel 1: C(bf16) = X(f32)[M,K] @ W(f32)[K,N]
//   128x128 block tile, 8 waves, each wave: 16 rows x 128 cols (8 accum tiles)
//   TRANS=1 stores C transposed (for V: Vt[d_v][n_kv])
// ---------------------------------------------------------------------------
template <int TRANS>
__global__ __launch_bounds__(256) void gemm_xw_bf16(
    const float* __restrict__ X, const float* __restrict__ W,
    uint16_t* __restrict__ Out, int M, int K, int N)
{
    // pitch 40 u16 (80B): 16B-aligned rows, conflict-free fragment reads
    __shared__ __align__(16) uint16_t Xs[128][40];   // A tile (row-major)
    __shared__ __align__(16) uint16_t Wt[128][40];   // B tile, [col][k]

    const int tid  = threadIdx.x;
    const int wid  = tid >> 5;
    const int lane = tid & 31;
    const int m16  = lane & 15;     // A row / B col within tile
    const int kh   = lane >> 4;     // K-half selector
    const int bRow = blockIdx.y * 128;
    const int bCol = blockIdx.x * 128;

    const v8f vzero = {0.f,0.f,0.f,0.f,0.f,0.f,0.f,0.f};
    v8f acc[8];
    #pragma unroll
    for (int i = 0; i < 8; ++i) acc[i] = vzero;

    for (int k0 = 0; k0 < K; k0 += 32) {
        // stage X tile 128x32 f32 -> bf16 (1024 float4 chunks / 256 threads)
        #pragma unroll
        for (int i = 0; i < 4; ++i) {
            int c  = tid + i * 256;
            int r  = c >> 3;
            int cp = (c & 7) * 4;
            const float4 v = *(const float4*)(X + (size_t)(bRow + r) * K + k0 + cp);
            *(uint2*)&Xs[r][cp] = make_uint2(f2bf2(v.x, v.y), f2bf2(v.z, v.w));
        }
        // stage W tile 32x128 f32 -> bf16, transposed into Wt[col][k]
        #pragma unroll
        for (int i = 0; i < 4; ++i) {
            int c  = tid + i * 256;
            int kk = c >> 5;
            int cp = (c & 31) * 4;
            const float4 v = *(const float4*)(W + (size_t)(k0 + kk) * N + bCol + cp);
            uint32_t p01 = f2bf2(v.x, v.y);
            uint32_t p23 = f2bf2(v.z, v.w);
            Wt[cp + 0][kk] = (uint16_t)p01;
            Wt[cp + 1][kk] = (uint16_t)(p01 >> 16);
            Wt[cp + 2][kk] = (uint16_t)p23;
            Wt[cp + 3][kk] = (uint16_t)(p23 >> 16);
        }
        __syncthreads();

        // A fragment: 16x32, lane row = m16; chunks at K = 8*kh and 16+8*kh
        AB32 a;
        {
            const uint16_t* row = &Xs[wid * 16 + m16][0];
            a.q[0] = *(const u32x4*)(row + kh * 8);
            a.q[1] = *(const u32x4*)(row + 16 + kh * 8);
        }
        #pragma unroll
        for (int ct = 0; ct < 8; ++ct) {
            // B fragment: 32x16, lane col = m16; chunks at K = 8*kh and 16+8*kh
            AB32 b;
            const uint16_t* row = &Wt[ct * 16 + m16][0];
            b.q[0] = *(const u32x4*)(row + kh * 8);
            b.q[1] = *(const u32x4*)(row + 16 + kh * 8);
            acc[ct] = WMMA_BF16(a.v, b.v, acc[ct]);
        }
        __syncthreads();
    }

    // C layout: VGPR j -> row (j + 8*(lane>>4)), col = lane&15.
    // Convert row pairs with one packed cvt, store 16-bit halves.
    #pragma unroll
    for (int ct = 0; ct < 8; ++ct) {
        #pragma unroll
        for (int j = 0; j < 8; j += 2) {
            int row0 = bRow + wid * 16 + j + 8 * kh;
            int col  = bCol + ct * 16 + m16;
            uint32_t p = f2bf2(acc[ct][j], acc[ct][j + 1]);
            if (TRANS) {
                Out[(size_t)col * M + row0]     = (uint16_t)p;
                Out[(size_t)col * M + row0 + 1] = (uint16_t)(p >> 16);
            } else {
                Out[(size_t)(row0)     * N + col] = (uint16_t)p;
                Out[(size_t)(row0 + 1) * N + col] = (uint16_t)(p >> 16);
            }
        }
    }
}

// ---------------------------------------------------------------------------
// Kernel 2: flash attention.
//   Qb[NQ][D], Kb[NKV][D] row-major bf16; Vt[D][NKV] bf16 (V transposed).
//   Block = 32 query rows, 8 waves. KV tile = 64.
//   Q staged once in LDS; K tile staged via async global->LDS copies that
//   overlap the softmax + P@V phase of the previous iteration (ASYNCcnt).
//   V fragments read directly from L2-resident global (no reuse in block).
// ---------------------------------------------------------------------------
#define AT_D     1024
#define AT_NKV   4096
#define AT_BQ    32
#define AT_BK    64
#define AT_QPIT  (AT_D + 8)     /* u16 pitch: 2064B rows, 16B aligned       */
#define AT_SCALE 0.03125f       /* 1/sqrt(1024) */

__global__ __launch_bounds__(256) void attn_flash_bf16(
    const uint16_t* __restrict__ Qb,
    const uint16_t* __restrict__ Kb,
    const uint16_t* __restrict__ Vt,
    float* __restrict__ Out)
{
    __shared__ __align__(16) uint16_t Qs[AT_BQ][AT_QPIT];       //  66.0 KB
    __shared__ __align__(16) uint16_t Ks[AT_BK][AT_QPIT];       // 132.1 KB
    __shared__ __align__(16) float    Sbuf[AT_BQ][AT_BK + 4];   //   8.7 KB
    __shared__ __align__(16) uint16_t Pbuf[AT_BQ][AT_BK + 8];   //   4.6 KB
    __shared__ float mS[AT_BQ], lS[AT_BQ], corrS[AT_BQ];

    const int tid  = threadIdx.x;
    const int wid  = tid >> 5;
    const int lane = tid & 31;
    const int m16  = lane & 15;
    const int kh   = lane >> 4;
    const int qBase = blockIdx.x * AT_BQ;

    const int s_qt = wid >> 2;      // S tile: q sub-tile 0..1
    const int s_kt = wid & 3;       // S tile: kv sub-tile 0..3

    const v8f vzero = {0.f,0.f,0.f,0.f,0.f,0.f,0.f,0.f};
    v8f oacc[16];                   // [qt*8 + ct]
    #pragma unroll
    for (int i = 0; i < 16; ++i) oacc[i] = vzero;

    if (tid < AT_BQ) { mS[tid] = -1e30f; lS[tid] = 0.f; }

    // ---- prologue: async-stage Q block (once) and first K tile --------------
    #pragma unroll
    for (int i = 0; i < 16; ++i) {                 // 32 rows x 128 chunks
        int c  = tid + i * 256;
        int r  = c >> 7;
        int cp = (c & 127) * 8;
        cp_async_b128(Qb + (size_t)(qBase + r) * AT_D + cp, &Qs[r][cp]);
    }
    #pragma unroll
    for (int i = 0; i < 32; ++i) {                 // 64 rows x 128 chunks
        int c  = tid + i * 256;
        int r  = c >> 7;
        int cp = (c & 127) * 8;
        cp_async_b128(Kb + (size_t)r * AT_D + cp, &Ks[r][cp]);
    }

    for (int kv0 = 0; kv0 < AT_NKV; kv0 += AT_BK) {
        wait_async0();             // my async copies into Qs/Ks are done
        __syncthreads();           // everyone's copies are done

        // ---------------- S = Q K^T (one 16x16 tile per wave) ----------------
        v8f sacc = vzero;
        const uint16_t* qrow = &Qs[s_qt * 16 + m16][0];
        const uint16_t* krow = &Ks[s_kt * 16 + m16][0];
        #pragma unroll 4
        for (int d0 = 0; d0 < AT_D; d0 += 32) {
            AB32 a, b;
            a.q[0] = *(const u32x4*)(qrow + d0 + kh * 8);
            a.q[1] = *(const u32x4*)(qrow + d0 + 16 + kh * 8);
            b.q[0] = *(const u32x4*)(krow + d0 + kh * 8);
            b.q[1] = *(const u32x4*)(krow + d0 + 16 + kh * 8);
            sacc = WMMA_BF16(a.v, b.v, sacc);
        }
        #pragma unroll
        for (int j = 0; j < 8; ++j)
            Sbuf[s_qt * 16 + j + 8 * kh][s_kt * 16 + m16] = sacc[j];
        __syncthreads();           // S complete; Ks no longer read this iter

        // ---- prefetch next K tile; overlaps softmax + P@V below -------------
        if (kv0 + AT_BK < AT_NKV) {
            const uint16_t* kNext = Kb + (size_t)(kv0 + AT_BK) * AT_D;
            #pragma unroll
            for (int i = 0; i < 32; ++i) {
                int c  = tid + i * 256;
                int r  = c >> 7;
                int cp = (c & 127) * 8;
                cp_async_b128(kNext + (size_t)r * AT_D + cp, &Ks[r][cp]);
            }
        }

        // ---------------- online softmax (wave 0, one row per lane) ----------
        if (wid == 0) {
            float mOld = mS[lane];
            float rmax = -1e30f;
            #pragma unroll
            for (int c = 0; c < AT_BK; ++c)
                rmax = fmaxf(rmax, Sbuf[lane][c]);
            float mNew = fmaxf(mOld, rmax * AT_SCALE);
            float corr = __expf(mOld - mNew);
            float lsum = 0.f;
            #pragma unroll
            for (int c = 0; c < AT_BK; c += 2) {
                float p0 = __expf(Sbuf[lane][c]     * AT_SCALE - mNew);
                float p1 = __expf(Sbuf[lane][c + 1] * AT_SCALE - mNew);
                lsum += p0 + p1;
                *(uint32_t*)&Pbuf[lane][c] = f2bf2(p0, p1);
            }
            mS[lane]    = mNew;
            lS[lane]    = lS[lane] * corr + lsum;
            corrS[lane] = corr;
        }
        __syncthreads();

        // ---------------- O = corr*O + P @ V ---------------------------------
        #pragma unroll
        for (int qt = 0; qt < 2; ++qt) {
            #pragma unroll
            for (int j = 0; j < 8; ++j) {
                float c = corrS[qt * 16 + j + 8 * kh];
                #pragma unroll
                for (int ct = 0; ct < 8; ++ct) oacc[qt * 8 + ct][j] *= c;
            }
        }
        #pragma unroll
        for (int ks = 0; ks < AT_BK / 32; ++ks) {
            AB32 a[2];
            #pragma unroll
            for (int qt = 0; qt < 2; ++qt) {
                const uint16_t* prow = &Pbuf[qt * 16 + m16][0];
                a[qt].q[0] = *(const u32x4*)(prow + ks * 32 + kh * 8);
                a[qt].q[1] = *(const u32x4*)(prow + ks * 32 + 16 + kh * 8);
            }
            #pragma unroll
            for (int ct = 0; ct < 8; ++ct) {
                const uint16_t* vrow = Vt + (size_t)(wid * 128 + ct * 16 + m16) * AT_NKV
                                          + kv0 + ks * 32;
                AB32 b;
                b.q[0] = *(const u32x4*)(vrow + kh * 8);
                b.q[1] = *(const u32x4*)(vrow + 16 + kh * 8);
                #pragma unroll
                for (int qt = 0; qt < 2; ++qt)
                    oacc[qt * 8 + ct] = WMMA_BF16(a[qt].v, b.v, oacc[qt * 8 + ct]);
            }
        }
        __syncthreads();           // Sbuf/Pbuf reused next iteration
    }

    // ---------------- epilogue: O / l -> f32 out -----------------------------
    #pragma unroll
    for (int qt = 0; qt < 2; ++qt) {
        #pragma unroll
        for (int j = 0; j < 8; ++j) {
            int   row = qt * 16 + j + 8 * kh;
            float inv = 1.0f / lS[row];
            float* orow = Out + (size_t)(qBase + row) * AT_D + wid * 128;
            #pragma unroll
            for (int ct = 0; ct < 8; ++ct)
                orow[ct * 16 + m16] = oacc[qt * 8 + ct][j] * inv;
        }
    }
}

// ---------------------------------------------------------------------------
// Launch: 3 projection GEMMs (Q, K row-major; V transposed) then attention.
// Workspace: Qb (8MB) | Kb (8MB) | Vt (8MB) bf16.
// ---------------------------------------------------------------------------
extern "C" void kernel_launch(void* const* d_in, const int* in_sizes, int n_in,
                              void* d_out, int out_size, void* d_ws, size_t ws_size,
                              hipStream_t stream) {
    const float* x1 = (const float*)d_in[0];   // [4096,1024]
    const float* x2 = (const float*)d_in[1];   // [4096,1024]
    const float* Wq = (const float*)d_in[2];   // [1024,1024]
    const float* Wk = (const float*)d_in[3];
    const float* Wv = (const float*)d_in[4];
    float* out = (float*)d_out;                // [4096,1024]

    uint16_t* Qb = (uint16_t*)d_ws;
    uint16_t* Kb = Qb + (size_t)4096 * 1024;
    uint16_t* Vt = Kb + (size_t)4096 * 1024;

    dim3 gproj(1024 / 128, 4096 / 128);
    gemm_xw_bf16<0><<<gproj, 256, 0, stream>>>(x1, Wq, Qb, 4096, 1024, 1024);
    gemm_xw_bf16<0><<<gproj, 256, 0, stream>>>(x2, Wk, Kb, 4096, 1024, 1024);
    gemm_xw_bf16<1><<<gproj, 256, 0, stream>>>(x2, Wv, Vt, 4096, 1024, 1024);

    attn_flash_bf16<<<4096 / AT_BQ, 256, 0, stream>>>(Qb, Kb, Vt, out);
}
